// MyGo_Loss3_83880711291626
// MI455X (gfx1250) — matrix operand order repaired and checked by reading, not compiled, not run on hardware
//
#include <hip/hip_runtime.h>
#include <hip/hip_bf16.h>

#define BBATCH 32
#define NN 784
#define MM 392
#define DD 1024
#define CC 20
#define KPROTO 4

typedef __attribute__((ext_vector_type(16))) _Float16 v16h;
typedef __attribute__((ext_vector_type(8)))  _Float16 v8h;
typedef __attribute__((ext_vector_type(8)))  float    v8f;

__device__ inline float bred256(float v, float* red) {
  int t = threadIdx.x;
  red[t] = v; __syncthreads();
  #pragma unroll
  for (int s = 128; s > 0; s >>= 1) {
    if (t < s) red[t] += red[t + s];
    __syncthreads();
  }
  float r = red[0]; __syncthreads();
  return r;
}

__device__ inline float wred32(float v) {
  #pragma unroll
  for (int off = 16; off > 0; off >>= 1) v += __shfl_xor(v, off, 32);
  return v;
}

// ---------------- Stage 1: per-point classification ----------------
__global__ void mygo_classify(const float* __restrict__ score, int* __restrict__ cls,
                              float* __restrict__ sm, int* __restrict__ cnt) {
  int idx = blockIdx.x * 256 + threadIdx.x;
  if (idx >= BBATCH * MM) return;
  int bb = idx / MM, mi = idx % MM;
  const float* s = score + ((size_t)bb * NN + 2 * mi) * CC;
  float best = s[0]; int bi = 0; float sum = s[0];
  #pragma unroll
  for (int k = 1; k < CC; k++) {
    float v = s[k]; sum += v;
    if (v > best) { best = v; bi = k; }   // first-max (jnp.argmax)
  }
  cls[idx] = bi;
  sm[idx] = sum * (1.0f / CC);
  atomicAdd(&cnt[bb * CC + bi], 1);
}

// ---------------- Stage 2: centroid init (stable argsort of ~sel) ----------------
__global__ void mygo_init(const float* __restrict__ query, const int* __restrict__ cls,
                          float* __restrict__ cen, float* __restrict__ cennorm) {
  __shared__ int sidx[3];
  __shared__ float red[256];
  int bc = blockIdx.x; int bb = bc / CC; int ci = bc % CC;
  int tid = threadIdx.x;
  if (tid == 0) {
    int t = 0;
    for (int mi = 0; mi < MM && t < 3; ++mi) if (cls[bb * MM + mi] == ci) sidx[t++] = mi;
    for (int mi = 0; mi < MM && t < 3; ++mi) if (cls[bb * MM + mi] != ci) sidx[t++] = mi;
  }
  __syncthreads();
  for (int jj = 0; jj < 3; jj++) {
    const float* src = query + ((size_t)bb * NN + 2 * sidx[jj]) * DD;
    float* dst = cen + ((size_t)(bc * 3 + jj)) * DD;
    float part = 0.f;
    #pragma unroll
    for (int k = 0; k < 4; k++) { int dd = tid + k * 256; float v = src[dd]; dst[dd] = v; part += v * v; }
    float tot = bred256(part, red);
    if (tid == 0) cennorm[bc * 3 + jj] = tot;
  }
}

// ---------------- k-means: assign (1 wave / point) ----------------
__global__ void mygo_assign(const float* __restrict__ query, const int* __restrict__ cls,
                            const float* __restrict__ cen, const float* __restrict__ cennorm,
                            int* __restrict__ kass) {
  int idx = blockIdx.x;
  int bb = idx / MM, mi = idx % MM;
  int lane = threadIdx.x;
  int ci = cls[idx];
  const float* x = query + ((size_t)bb * NN + 2 * mi) * DD;
  const float* ce = cen + ((size_t)((bb * CC + ci) * 3)) * DD;
  float d0 = 0.f, d1 = 0.f, d2 = 0.f;
  #pragma unroll 4
  for (int e = 0; e < DD / 32; e++) {
    int dd = lane + 32 * e;
    float xv = x[dd];
    d0 += xv * ce[dd]; d1 += xv * ce[DD + dd]; d2 += xv * ce[2 * DD + dd];
  }
  d0 = wred32(d0); d1 = wred32(d1); d2 = wred32(d2);
  if (lane == 0) {
    int base = (bb * CC + ci) * 3;
    float v0 = cennorm[base + 0] - 2.f * d0;
    float v1 = cennorm[base + 1] - 2.f * d1;
    float v2 = cennorm[base + 2] - 2.f * d2;
    int bj = 0; float bv = v0;
    if (v1 < bv) { bv = v1; bj = 1; }
    if (v2 < bv) { bv = v2; bj = 2; }
    kass[idx] = bj;
  }
}

// ---------------- k-means: accumulate sums + counts (block per (b,c,j)) ----------------
__global__ void mygo_accum(const float* __restrict__ query, const int* __restrict__ cls,
                           const int* __restrict__ kass, float* __restrict__ sums,
                           int* __restrict__ kcnt) {
  __shared__ int scls[MM]; __shared__ int skas[MM];
  int id = blockIdx.x; int j = id % 3; int bc = id / 3; int bb = bc / CC; int ci = bc % CC;
  int tid = threadIdx.x;
  for (int t = tid; t < MM; t += 256) { scls[t] = cls[bb * MM + t]; skas[t] = kass[bb * MM + t]; }
  __syncthreads();
  float a0 = 0.f, a1 = 0.f, a2 = 0.f, a3 = 0.f; int cntl = 0;
  for (int mi = 0; mi < MM; ++mi) {
    if (scls[mi] == ci && skas[mi] == j) {
      const float* x = query + ((size_t)bb * NN + 2 * mi) * DD;
      a0 += x[tid]; a1 += x[tid + 256]; a2 += x[tid + 512]; a3 += x[tid + 768];
      cntl++;
    }
  }
  float* dst = sums + (size_t)id * DD;
  dst[tid] = a0; dst[tid + 256] = a1; dst[tid + 512] = a2; dst[tid + 768] = a3;
  if (tid == 0) kcnt[id] = cntl;
}

// ---------------- k-means: update centroids + norms ----------------
__global__ void mygo_update(const float* __restrict__ sums, const int* __restrict__ kcnt,
                            float* __restrict__ cen, float* __restrict__ cennorm) {
  __shared__ float red[256];
  int id = blockIdx.x; int tid = threadIdx.x;
  int kc = kcnt[id];
  float inv = (kc > 0) ? 1.0f / (float)kc : 0.f;
  float part = 0.f;
  #pragma unroll
  for (int k = 0; k < 4; k++) {
    int dd = tid + k * 256;
    float v;
    if (kc > 0) { v = sums[(size_t)id * DD + dd] * inv; cen[(size_t)id * DD + dd] = v; }
    else        { v = cen[(size_t)id * DD + dd]; }
    part += v * v;
  }
  float tot = bred256(part, red);
  if (tid == 0) cennorm[id] = tot;
}

// ---------------- avg score per cluster + stable descending sort ----------------
__global__ void mygo_avg(const int* __restrict__ cls, const int* __restrict__ kass,
                         const float* __restrict__ sm, int* __restrict__ ordj) {
  __shared__ float red[256];
  int bc = blockIdx.x; int bb = bc / CC; int ci = bc % CC; int tid = threadIdx.x;
  float s0 = 0.f, s1 = 0.f, s2 = 0.f, c0 = 0.f, c1 = 0.f, c2 = 0.f;
  for (int mi = tid; mi < MM; mi += 256) {
    if (cls[bb * MM + mi] == ci) {
      int j = kass[bb * MM + mi]; float v = sm[bb * MM + mi];
      if (j == 0) { s0 += v; c0 += 1.f; }
      else if (j == 1) { s1 += v; c1 += 1.f; }
      else { s2 += v; c2 += 1.f; }
    }
  }
  s0 = bred256(s0, red); s1 = bred256(s1, red); s2 = bred256(s2, red);
  c0 = bred256(c0, red); c1 = bred256(c1, red); c2 = bred256(c2, red);
  if (tid == 0) {
    float ninf = -__builtin_inff();
    float a0 = (c0 > 0.f) ? s0 / c0 : ninf;
    float a1 = (c1 > 0.f) ? s1 / c1 : ninf;
    float a2 = (c2 > 0.f) ? s2 / c2 : ninf;
    int o0 = 0, o1 = 1, o2 = 2; float t; int oi;
    if (a1 > a0) { t = a0; a0 = a1; a1 = t; oi = o0; o0 = o1; o1 = oi; }
    if (a2 > a1) { t = a1; a1 = a2; a2 = t; oi = o1; o1 = o2; o2 = oi; }
    if (a1 > a0) { t = a0; a0 = a1; a1 = t; oi = o0; o0 = o1; o1 = oi; }
    ordj[bc * 3 + 0] = o0; ordj[bc * 3 + 1] = o1; ordj[bc * 3 + 2] = o2;
  }
}

// ---------------- per-class member sums (for backbone) ----------------
__global__ void mygo_classsum(const float* __restrict__ query, const int* __restrict__ cls,
                              float* __restrict__ classsum) {
  __shared__ int scls[MM];
  int bc = blockIdx.x; int bb = bc / CC; int ci = bc % CC; int tid = threadIdx.x;
  for (int t = tid; t < MM; t += 256) scls[t] = cls[bb * MM + t];
  __syncthreads();
  float a0 = 0.f, a1 = 0.f, a2 = 0.f, a3 = 0.f;
  for (int mi = 0; mi < MM; ++mi) {
    if (scls[mi] == ci) {
      const float* x = query + ((size_t)bb * NN + 2 * mi) * DD;
      a0 += x[tid]; a1 += x[tid + 256]; a2 += x[tid + 512]; a3 += x[tid + 768];
    }
  }
  float* dst = classsum + (size_t)bc * DD;
  dst[tid] = a0; dst[tid + 256] = a1; dst[tid + 512] = a2; dst[tid + 768] = a3;
}

__global__ void mygo_totalsum(const float* __restrict__ classsum, float* __restrict__ totalsum) {
  int idx = blockIdx.x * 256 + threadIdx.x;   // BBATCH*DD
  int bb = idx / DD, dd = idx % DD;
  float s = 0.f;
  #pragma unroll
  for (int ci = 0; ci < CC; ++ci) s += classsum[((size_t)(bb * CC + ci)) * DD + dd];
  totalsum[idx] = s;
}

// ---------------- build + L2-normalize protos, store f16 ----------------
__global__ void mygo_protos(const float* __restrict__ cen, const int* __restrict__ ordj,
                            const float* __restrict__ classsum, const float* __restrict__ totalsum,
                            const int* __restrict__ cnt, _Float16* __restrict__ ph16) {
  __shared__ float red[256];
  int id = blockIdx.x; int jj = id & 3; int bc = id >> 2; int bb = bc / CC;
  int tid = threadIdx.x;
  float vv[4]; float part = 0.f;
  if (jj < 3) {
    int js = ordj[bc * 3 + jj];
    const float* src = cen + ((size_t)(bc * 3 + js)) * DD;
    #pragma unroll
    for (int k = 0; k < 4; k++) { int dd = tid + k * 256; float v = src[dd]; vv[k] = v; part += v * v; }
  } else {
    float denom = fmaxf((float)(MM - cnt[bc]), 1.0f);
    const float* ts = totalsum + (size_t)bb * DD;
    const float* cs = classsum + (size_t)bc * DD;
    #pragma unroll
    for (int k = 0; k < 4; k++) { int dd = tid + k * 256; float v = (ts[dd] - cs[dd]) / denom; vv[k] = v; part += v * v; }
  }
  float tot = bred256(part, red);
  float invn = 1.0f / fmaxf(sqrtf(tot), 1e-12f);
  #pragma unroll
  for (int k = 0; k < 4; k++) { int dd = tid + k * 256; ph16[(size_t)id * DD + dd] = (_Float16)(vv[k] * invn); }
}

// ---------------- normalize query[0], store f16 ----------------
__global__ void mygo_q0h(const float* __restrict__ query, _Float16* __restrict__ q0h) {
  __shared__ float red[256];
  int i = blockIdx.x; int tid = threadIdx.x;
  const float* src = query + (size_t)i * DD;     // b = 0 rows, full n
  float vv[4]; float part = 0.f;
  #pragma unroll
  for (int k = 0; k < 4; k++) { int dd = tid + k * 256; float v = src[dd]; vv[k] = v; part += v * v; }
  float tot = bred256(part, red);
  float invn = 1.0f / fmaxf(sqrtf(tot), 1e-12f);
  #pragma unroll
  for (int k = 0; k < 4; k++) { int dd = tid + k * 256; q0h[(size_t)i * DD + dd] = (_Float16)(vv[k] * invn); }
}

// ---------------- WMMA: sim = q0h * ph^T, Km = exp((sim-1)/eps) ----------------
// One wave per (b, class-quad, 16-row n-tile). N=16 cols = 4 classes x 4 protos (no padding waste).
__global__ void mygo_km(const _Float16* __restrict__ q0h, const _Float16* __restrict__ ph,
                        float* __restrict__ Km) {
  int id = blockIdx.x;
  int mt = id % 49; int cq = (id / 49) % 5; int bb = id / 245;
  int lane = threadIdx.x;
  int r16 = lane & 15;
  bool hi = lane >= 16;
  const _Float16* Ap = q0h + (size_t)(mt * 16 + r16) * DD;        // A row M = r16
  const _Float16* Bp = ph + ((size_t)bb * (CC * 4) + cq * 16 + r16) * DD;  // B col N = r16
  const int aoff0 = hi ? 8 : 0;    // A: lanes 0-15 hold K 0-7/16-23; lanes 16-31 hold K 8-15/24-31
  const int boff0 = hi ? 8 : 0;    // B: lanes 0-15 hold K 0-7/16-23; lanes 16-31 hold K 8-15/24-31
  v8f acc = {};
  for (int k0 = 0; k0 < DD; k0 += 32) {
    v8h alo = *(const v8h*)(Ap + k0 + aoff0);
    v8h ahi = *(const v8h*)(Ap + k0 + aoff0 + 16);
    v8h blo = *(const v8h*)(Bp + k0 + boff0);
    v8h bhi = *(const v8h*)(Bp + k0 + boff0 + 16);
    v16h a = __builtin_shufflevector(alo, ahi, 0,1,2,3,4,5,6,7,8,9,10,11,12,13,14,15);
    v16h b = __builtin_shufflevector(blo, bhi, 0,1,2,3,4,5,6,7,8,9,10,11,12,13,14,15);
    acc = __builtin_amdgcn_wmma_f32_16x16x32_f16(false, a, false, b, (short)0, acc, false, false);
  }
  int ci = cq * 4 + (r16 >> 2);
  int j = r16 & 3;
  float* base = Km + (((size_t)(bb * CC + ci)) * NN) * 4 + j;
  #pragma unroll
  for (int v = 0; v < 8; ++v) {
    int r = mt * 16 + v + (hi ? 8 : 0);
    base[(size_t)r * 4] = __expf(10.0f * (acc[v] - 1.0f));   // exp(-(1-sim)/0.1)
  }
}

// ---------------- Sinkhorn + T + pred + BCE, block per (b,c) ----------------
__global__ void mygo_sink(const float* __restrict__ Km, const int* __restrict__ gt,
                          const int* __restrict__ label, const int* __restrict__ cnt,
                          const float* __restrict__ wts, float* __restrict__ acc) {
  __shared__ float red[256];
  int bc = blockIdx.x; int bb = bc / CC; int ci = bc % CC; int tid = threadIdx.x;
  const float* kp = Km + (size_t)bc * NN * 4;
  float kr[4][4]; float rv[4];
  #pragma unroll
  for (int s = 0; s < 4; s++) {
    int i = tid + s * 256;
    rv[s] = 1.f;
    if (i < NN) {
      kr[s][0] = kp[i * 4 + 0]; kr[s][1] = kp[i * 4 + 1];
      kr[s][2] = kp[i * 4 + 2]; kr[s][3] = kp[i * 4 + 3];
    } else { kr[s][0] = kr[s][1] = kr[s][2] = kr[s][3] = 0.f; }
  }
  float cc0 = 1.f, cc1 = 1.f, cc2 = 1.f, cc3 = 1.f;
  const float uconst = 1.0f / NN;
  for (int it = 0; it < 100; ++it) {
    float rn[4]; float p0 = 0.f, p1 = 0.f, p2 = 0.f, p3 = 0.f, ep = 0.f;
    #pragma unroll
    for (int s = 0; s < 4; s++) {
      int i = tid + s * 256;
      if (i < NN) {
        float den = kr[s][0] * cc0 + kr[s][1] * cc1 + kr[s][2] * cc2 + kr[s][3] * cc3;
        float r1 = uconst / den;
        ep += fabsf(r1 - rv[s]);
        rn[s] = r1;
        p0 += kr[s][0] * r1; p1 += kr[s][1] * r1; p2 += kr[s][2] * r1; p3 += kr[s][3] * r1;
      } else rn[s] = rv[s];
    }
    float esum = bred256(ep, red);
    p0 = bred256(p0, red); p1 = bred256(p1, red); p2 = bred256(p2, red); p3 = bred256(p3, red);
    cc0 = 0.25f / p0; cc1 = 0.25f / p1; cc2 = 0.25f / p2; cc3 = 0.25f / p3;
    #pragma unroll
    for (int s = 0; s < 4; s++) rv[s] = rn[s];
    if (esum * (1.0f / NN) < 0.01f) break;   // freeze after updating r/cc (matches ref)
  }
  float w0 = wts[0], w1 = wts[1], w2 = wts[2], w3 = wts[3];
  float bcep = 0.f, nanp = 0.f;
  #pragma unroll
  for (int s = 0; s < 4; s++) {
    int i = tid + s * 256;
    if (i < NN) {
      float t0 = rv[s] * cc0 * kr[s][0], t1 = rv[s] * cc1 * kr[s][1];
      float t2 = rv[s] * cc2 * kr[s][2], t3 = rv[s] * cc3 * kr[s][3];
      if (t0 != t0 || t1 != t1 || t2 != t2 || t3 != t3) nanp = 1.f;
      float pred = t0 * w0 + t1 * w1 + t2 * w2 + t3 * w3;
      float p = fminf(fmaxf(pred, 0.f), 1.f);
      int y = (gt[bb * NN + i] == ci + 1) ? 1 : 0;
      float lp = fmaxf(logf(p), -100.f);
      float lq = fmaxf(logf(1.f - p), -100.f);
      bcep += -(y ? lp : lq);
    }
  }
  float bces = bred256(bcep, red);
  float nans = bred256(nanp, red);
  if (tid == 0) {
    bool valid = (label[bc] > 0) && (cnt[bc] >= KPROTO) && (nans == 0.f);
    if (valid) { atomicAdd(&acc[0], bces * (1.0f / NN)); atomicAdd(&acc[1], 1.0f); }
  }
}

__global__ void mygo_fin(const float* __restrict__ acc, float* __restrict__ out) {
  out[0] = acc[0] / (acc[1] + 1e-4f);
}

extern "C" void kernel_launch(void* const* d_in, const int* in_sizes, int n_in,
                              void* d_out, int out_size, void* d_ws, size_t ws_size,
                              hipStream_t stream) {
  (void)in_sizes; (void)n_in; (void)out_size; (void)ws_size;
  const float* query = (const float*)d_in[0];
  const float* score = (const float*)d_in[1];
  const int*   label = (const int*)d_in[2];
  const int*   gt    = (const int*)d_in[3];
  const float* wts   = (const float*)d_in[4];
  float* out = (float*)d_out;

  char* w = (char*)d_ws;
  auto carve = [&](size_t bytes) -> void* {
    void* p = (void*)w;
    w += (bytes + 255) & ~(size_t)255;
    return p;
  };
  int*      cls      = (int*)carve((size_t)BBATCH * MM * 4);
  int*      kass     = (int*)carve((size_t)BBATCH * MM * 4);
  int*      cnt      = (int*)carve((size_t)BBATCH * CC * 4);
  float*    sm       = (float*)carve((size_t)BBATCH * MM * 4);
  float*    cen      = (float*)carve((size_t)BBATCH * CC * 3 * DD * 4);
  float*    cennorm  = (float*)carve((size_t)BBATCH * CC * 3 * 4);
  float*    sums     = (float*)carve((size_t)BBATCH * CC * 3 * DD * 4);
  int*      kcnt     = (int*)carve((size_t)BBATCH * CC * 3 * 4);
  int*      ordj     = (int*)carve((size_t)BBATCH * CC * 3 * 4);
  float*    classsum = (float*)carve((size_t)BBATCH * CC * DD * 4);
  float*    totalsum = (float*)carve((size_t)BBATCH * DD * 4);
  _Float16* ph16     = (_Float16*)carve((size_t)BBATCH * CC * 4 * DD * 2);
  _Float16* q0h16    = (_Float16*)carve((size_t)NN * DD * 2);
  float*    Kmat     = (float*)carve((size_t)BBATCH * CC * NN * 4 * 4);
  float*    accum    = (float*)carve(2 * 4);

  hipMemsetAsync(cnt, 0, (size_t)BBATCH * CC * 4, stream);
  hipMemsetAsync(accum, 0, 8, stream);

  mygo_classify<<<(BBATCH * MM + 255) / 256, 256, 0, stream>>>(score, cls, sm, cnt);
  mygo_init<<<BBATCH * CC, 256, 0, stream>>>(query, cls, cen, cennorm);

  for (int it = 0; it < 10; ++it) {
    mygo_assign<<<BBATCH * MM, 32, 0, stream>>>(query, cls, cen, cennorm, kass);
    mygo_accum<<<BBATCH * CC * 3, 256, 0, stream>>>(query, cls, kass, sums, kcnt);
    mygo_update<<<BBATCH * CC * 3, 256, 0, stream>>>(sums, kcnt, cen, cennorm);
  }
  mygo_assign<<<BBATCH * MM, 32, 0, stream>>>(query, cls, cen, cennorm, kass);
  mygo_avg<<<BBATCH * CC, 256, 0, stream>>>(cls, kass, sm, ordj);

  mygo_classsum<<<BBATCH * CC, 256, 0, stream>>>(query, cls, classsum);
  mygo_totalsum<<<(BBATCH * DD) / 256, 256, 0, stream>>>(classsum, totalsum);
  mygo_protos<<<BBATCH * CC * 4, 256, 0, stream>>>(cen, ordj, classsum, totalsum, cnt, ph16);
  mygo_q0h<<<NN, 256, 0, stream>>>(query, q0h16);

  mygo_km<<<BBATCH * 5 * 49, 32, 0, stream>>>(q0h16, ph16, Kmat);

  mygo_sink<<<BBATCH * CC, 256, 0, stream>>>(Kmat, gt, label, cnt, wts, accum);
  mygo_fin<<<1, 1, 0, stream>>>(accum, out);
}
